// SPGATNet_55233279426720
// MI455X (gfx1250) — matrix operand order repaired and checked by reading, not compiled, not run on hardware
//
#include <hip/hip_runtime.h>
#include <hip/hip_bf16.h>

// ---------------------------------------------------------------------------
// 2-layer GAT (H=2, C=128, concat=False) + sigmoid edge score, for gfx1250.
// GEMM runs on v_wmma_f32_16x16x32_bf16 (fp32 accumulate); everything else is
// the memory/atomic-bound graph part (gather + segment softmax + scatter-add).
// ---------------------------------------------------------------------------

typedef __attribute__((ext_vector_type(16))) __bf16 v16bf;
typedef __attribute__((ext_vector_type(8)))  float  v8f;

#define C_DIM 128
#define HC    256   // H * C

// float -> bf16 (round to nearest even), via raw bits
__device__ __forceinline__ unsigned short f2bf_raw(float f) {
    unsigned u = __builtin_bit_cast(unsigned, f);
    unsigned r = u + 0x7FFFu + ((u >> 16) & 1u);
    return (unsigned short)(r >> 16);
}

// monotonic float<->uint encoding so unsigned atomicMax == float max
__device__ __forceinline__ unsigned enc_f32(float f) {
    unsigned u = __builtin_bit_cast(unsigned, f);
    return (u & 0x80000000u) ? ~u : (u | 0x80000000u);
}
__device__ __forceinline__ float dec_f32(unsigned u) {
    unsigned b = (u & 0x80000000u) ? (u & 0x7FFFFFFFu) : ~u;
    return __builtin_bit_cast(float, b);
}

// ---------------------------------------------------------------------------
__global__ void k_zero(unsigned* __restrict__ p, long long count) {
    long long t = (long long)blockIdx.x * blockDim.x + threadIdx.x;
    if (t < count) p[t] = 0u;
}

__global__ void k_cast_x(const float* __restrict__ x,
                         unsigned short* __restrict__ xb, int tot) {
    int t = blockIdx.x * blockDim.x + threadIdx.x;
    if (t < tot) xb[t] = f2bf_raw(x[t]);
}

// W is [128, 256] row-major; produce WT[ncol][k] (bf16) so B-fragment loads
// are contiguous along K.
__global__ void k_cast_w(const float* __restrict__ W,
                         unsigned short* __restrict__ wt) {
    int t = blockIdx.x * blockDim.x + threadIdx.x;
    if (t >= HC * C_DIM) return;
    int ncol = t >> 7, k = t & 127;
    wt[t] = f2bf_raw(W[k * HC + ncol]);
}

// ---------------------------------------------------------------------------
// XP[N,256] = Xb[N,128] @ Wt^T.
// One wave computes a 16x64 output strip: the A fragment (16x32 per K-chunk)
// is loaded once and reused across 4 B fragments / 4 accumulators
// (16 v_wmma per wave). 4 waves/block cover all 16 N-tiles; grid.x = M tiles.
//
// A (16x32 bf16) ISA layout: lanes 0-15 = M0-15 {K 0..7, 16..23},
//                            lanes 16-31 = M0-15 {K 8..15, 24..31}.
// B (32x16 bf16) layout: lanes 0-15 = N0-15 {K 0..7, 16..23}, lanes 16-31 {K 8..15, 24..31}.
// C/D: lanes 0-15 N=lane M=r, lanes 16-31 N=lane-16 M=8+r.
__global__ void k_gemm(const unsigned short* __restrict__ xb,
                       const unsigned short* __restrict__ wt,
                       float* __restrict__ xp, int nrows) {
    const int lane = threadIdx.x & 31;
    const int wv   = threadIdx.x >> 5;       // 4 waves/block
    const int tM   = blockIdx.x;
    const int tN0  = wv * 4;                 // this wave's first N tile (of 16)
    const int half = lane >> 4;
    const int m16  = lane & 15;

    int row = tM * 16 + m16;
    const bool full = (tM * 16 + 16) <= nrows;   // wave-uniform
    if (row >= nrows) row = nrows - 1;           // clamp, keep EXEC all-ones

    v8f acc[4] = {};
#pragma unroll
    for (int kk = 0; kk < C_DIM; kk += 32) {
        v16bf a;
        const unsigned short* ap = xb + row * C_DIM + kk + half * 8;
#pragma unroll
        for (int i = 0; i < 8; ++i) {            // two contiguous 16B chunks
            a[i]     = __builtin_bit_cast(__bf16, ap[i]);
            a[i + 8] = __builtin_bit_cast(__bf16, ap[i + 16]);
        }
#pragma unroll
        for (int j = 0; j < 4; ++j) {            // reuse A across 4 N tiles
            v16bf b;
            const unsigned short* bp =
                wt + ((tN0 + j) * 16 + m16) * C_DIM + kk + half * 8;
#pragma unroll
            for (int i = 0; i < 8; ++i) {        // two contiguous 16B chunks
                b[i]     = __builtin_bit_cast(__bf16, bp[i]);
                b[i + 8] = __builtin_bit_cast(__bf16, bp[i + 16]);
            }
            acc[j] = __builtin_amdgcn_wmma_f32_16x16x32_bf16(
                false, a, false, b, (short)0, acc[j], false, false);
        }
    }

    if (full) {                                  // common case: no guards
#pragma unroll
        for (int j = 0; j < 4; ++j)
#pragma unroll
            for (int r = 0; r < 8; ++r)
                xp[(tM * 16 + half * 8 + r) * HC + (tN0 + j) * 16 + m16] =
                    acc[j][r];
    } else {
#pragma unroll
        for (int j = 0; j < 4; ++j)
#pragma unroll
            for (int r = 0; r < 8; ++r) {
                int orow = tM * 16 + half * 8 + r;
                if (orow < nrows)
                    xp[orow * HC + (tN0 + j) * 16 + m16] = acc[j][r];
            }
    }
}

// ---------------------------------------------------------------------------
// a_i[n,h] = dot(xp[n,h,:], a_src[h]);  a_j likewise. One wave per node.
__global__ void k_att(const float* __restrict__ xp,
                      const float* __restrict__ as_,
                      const float* __restrict__ ad_,
                      float* __restrict__ ai, float* __restrict__ aj, int n) {
    int node = blockIdx.x * 8 + (threadIdx.x >> 5);
    int lane = threadIdx.x & 31;
    if (node >= n) return;
#pragma unroll
    for (int h = 0; h < 2; ++h) {
        float si = 0.f, sj = 0.f;
        const float* xr = xp + node * HC + h * C_DIM;
#pragma unroll
        for (int j = 0; j < 4; ++j) {
            int c = lane + 32 * j;
            float v = xr[c];
            si += v * as_[h * C_DIM + c];
            sj += v * ad_[h * C_DIM + c];
        }
#pragma unroll
        for (int o = 16; o > 0; o >>= 1) {
            si += __shfl_xor(si, o, 32);
            sj += __shfl_xor(sj, o, 32);
        }
        if (lane == 0) { ai[node * 2 + h] = si; aj[node * 2 + h] = sj; }
    }
}

// ---------------------------------------------------------------------------
// Segment max of leaky_relu(a_i[src]+a_j[dst], 0.2). One thread per (edge,h).
// Self loops are edges E..E+n-1. AMAX pre-zeroed (encoded identity).
__global__ void k_edge_max(const int* __restrict__ src, const int* __restrict__ dst,
                           const float* __restrict__ ai, const float* __restrict__ aj,
                           unsigned* __restrict__ amax, int E, int n) {
    int t = blockIdx.x * blockDim.x + threadIdx.x;
    if (t >= (E + n) * 2) return;
    int e = t >> 1, h = t & 1, s, d;
    if (e < E) { s = src[e]; d = dst[e]; } else { s = d = e - E; }
    float v = ai[s * 2 + h] + aj[d * 2 + h];
    v = v > 0.f ? v : 0.2f * v;
    atomicMax(&amax[d * 2 + h], enc_f32(v));
}

__global__ void k_edge_exp(const int* __restrict__ src, const int* __restrict__ dst,
                           const float* __restrict__ ai, const float* __restrict__ aj,
                           const unsigned* __restrict__ amax,
                           float* __restrict__ exb, float* __restrict__ den,
                           int E, int n) {
    int t = blockIdx.x * blockDim.x + threadIdx.x;
    if (t >= (E + n) * 2) return;
    int e = t >> 1, h = t & 1, s, d;
    if (e < E) { s = src[e]; d = dst[e]; } else { s = d = e - E; }
    float v = ai[s * 2 + h] + aj[d * 2 + h];
    v = v > 0.f ? v : 0.2f * v;
    float ex = __expf(v - dec_f32(amax[d * 2 + h]));
    exb[e * 2 + h] = ex;
    atomicAdd(&den[d * 2 + h], ex);
}

// ---------------------------------------------------------------------------
// acc[dst,h,c] += att * xp[src,h,c]. One wave per edge; lane covers 8
// contiguous floats (wave covers the full 1KB row -> coalesced gather).
__global__ void k_agg(const int* __restrict__ src, const int* __restrict__ dst,
                      const float* __restrict__ exb, const float* __restrict__ den,
                      const float* __restrict__ xp, float* __restrict__ acc,
                      int E, int n) {
    int e = blockIdx.x * 8 + (threadIdx.x >> 5);
    if (e >= E + n) return;
    int lane = threadIdx.x & 31, s, d;
    if (e < E) { s = src[e]; d = dst[e]; } else { s = d = e - E; }
    int h = lane >> 4;                            // cols 0-127 -> h0, rest h1
    float att = exb[e * 2 + h] / (den[d * 2 + h] + 1e-16f);
    const float* xr = xp + (long long)s * HC + lane * 8;
    float* orow     = acc + (long long)d * HC + lane * 8;
    __builtin_prefetch(xr, 0, 1);                 // global_prefetch_b8
#pragma unroll
    for (int i = 0; i < 8; ++i)
        atomicAdd(&orow[i], att * xr[i]);
}

// ---------------------------------------------------------------------------
// mean over heads + bias -> xpre (kept for edge score), leaky(0.01) -> xout
__global__ void k_fin(const float* __restrict__ acc, const float* __restrict__ bias,
                      float* __restrict__ xpre, float* __restrict__ xout, int n) {
    int t = blockIdx.x * blockDim.x + threadIdx.x;
    if (t >= n * C_DIM) return;
    int node = t >> 7, c = t & 127;
    float pre = 0.5f * (acc[node * HC + c] + acc[node * HC + C_DIM + c]) + bias[c];
    xpre[t] = pre;
    xout[t] = pre > 0.f ? pre : 0.01f * pre;
}

// score[e] = sigmoid(sum_c xpre[src0]*xpre[dst0]*edge_attr). One wave/edge.
__global__ void k_score(const int* __restrict__ src, const int* __restrict__ dst,
                        const float* __restrict__ xpre, const float* __restrict__ ea,
                        float* __restrict__ sc, int E) {
    int e = blockIdx.x * 8 + (threadIdx.x >> 5);
    if (e >= E) return;
    int lane = threadIdx.x & 31;
    int s = src[e], d = dst[e];
    float a = 0.f;
#pragma unroll
    for (int j = 0; j < 4; ++j) {
        int c = lane + 32 * j;
        a += xpre[s * C_DIM + c] * xpre[d * C_DIM + c] * ea[(long long)e * C_DIM + c];
    }
#pragma unroll
    for (int o = 16; o > 0; o >>= 1) a += __shfl_xor(a, o, 32);
    if (lane == 0) sc[e] = 1.f / (1.f + __expf(-a));
}

// ---------------------------------------------------------------------------
extern "C" void kernel_launch(void* const* d_in, const int* in_sizes, int n_in,
                              void* d_out, int out_size, void* d_ws, size_t ws_size,
                              hipStream_t stream) {
    const float* x0 = (const float*)d_in[0];
    const int*   ei = (const int*)d_in[1];
    const float* ea = (const float*)d_in[2];
    const float* W[2]  = {(const float*)d_in[3], (const float*)d_in[7]};
    const float* As[2] = {(const float*)d_in[4], (const float*)d_in[8]};
    const float* Ad[2] = {(const float*)d_in[5], (const float*)d_in[9]};
    const float* Bs[2] = {(const float*)d_in[6], (const float*)d_in[10]};

    const int n  = in_sizes[0] / C_DIM;   // 50000
    const int E  = in_sizes[1] / 2;       // 800000
    const int Ep = E + n;                 // with self loops
    const int* src = ei;
    const int* dst = ei + E;

    // bump allocator over d_ws
    char* p = (char*)d_ws;
    auto alloc = [&](size_t bytes) -> char* {
        char* r = p; p += (bytes + 255) & ~(size_t)255; return r;
    };
    float*          XP   = (float*)         alloc((size_t)n * HC * 4);
    float*          ACC  = (float*)         alloc((size_t)n * HC * 4);
    unsigned short* XB   = (unsigned short*)alloc((size_t)n * C_DIM * 2);
    unsigned short* WT   = (unsigned short*)alloc((size_t)HC * C_DIM * 2);
    float*          AI   = (float*)         alloc((size_t)n * 2 * 4);
    float*          AJ   = (float*)         alloc((size_t)n * 2 * 4);
    unsigned*       AMAX = (unsigned*)      alloc((size_t)n * 2 * 4);
    float*          DEN  = (float*)         alloc((size_t)n * 2 * 4);
    float*          EXB  = (float*)         alloc((size_t)Ep * 2 * 4);
    float*          XPRE = (float*)         alloc((size_t)n * C_DIM * 4);
    float*          XCUR = (float*)         alloc((size_t)n * C_DIM * 4);

    float* xout_final = (float*)d_out;                       // [n*128]
    float* score_out  = xout_final + (size_t)n * C_DIM;      // [E]

    const int Mtiles = (n + 15) / 16;
    const int totx   = n * C_DIM;

    const float* xin = x0;
    for (int L = 0; L < 2; ++L) {
        // casts
        k_cast_x<<<(totx + 255) / 256, 256, 0, stream>>>(xin, XB, totx);
        k_cast_w<<<(HC * C_DIM + 255) / 256, 256, 0, stream>>>(W[L], WT);
        // zero accumulators (graph-capture-safe; no memset API)
        k_zero<<<(n * 2 + 255) / 256, 256, 0, stream>>>(AMAX, (long long)n * 2);
        k_zero<<<(n * 2 + 255) / 256, 256, 0, stream>>>((unsigned*)DEN, (long long)n * 2);
        k_zero<<<((long long)n * HC + 255) / 256, 256, 0, stream>>>((unsigned*)ACC, (long long)n * HC);
        // feature transform (WMMA): 1 block per M tile, 4 waves x 4 N-tiles
        k_gemm<<<Mtiles, 128, 0, stream>>>(XB, WT, XP, n);
        // per-node attention coefficients
        k_att<<<(n + 7) / 8, 256, 0, stream>>>(XP, As[L], Ad[L], AI, AJ, n);
        // segment softmax
        int tot2 = Ep * 2;
        k_edge_max<<<(tot2 + 255) / 256, 256, 0, stream>>>(src, dst, AI, AJ, AMAX, E, n);
        k_edge_exp<<<(tot2 + 255) / 256, 256, 0, stream>>>(src, dst, AI, AJ, AMAX, EXB, DEN, E, n);
        // message aggregation
        k_agg<<<(Ep + 7) / 8, 256, 0, stream>>>(src, dst, EXB, DEN, XP, ACC, E, n);
        // finalize: mean heads + bias, leaky-relu
        float* xpost = (L == 0) ? XCUR : xout_final;
        k_fin<<<(totx + 255) / 256, 256, 0, stream>>>(ACC, Bs[L], XPRE, xpost, n);
        xin = XCUR;
    }
    // edge score from layer-2 pre-activation features
    k_score<<<(E + 7) / 8, 256, 0, stream>>>(src, dst, XPRE, ea, score_out, E);
}
